// ref_network_23716809408856
// MI455X (gfx1250) — matrix operand order repaired and checked by reading, not compiled, not run on hardware
//
#include <hip/hip_runtime.h>

typedef _Float16 f16;
typedef __attribute__((ext_vector_type(2)))  _Float16 h2;    // (hi, lo) pair
typedef __attribute__((ext_vector_type(16))) _Float16 v16h;
typedef __attribute__((ext_vector_type(8)))  float    v8f;

#define TPB 256

// ---------------------------------------------------------------------------
// Weight prep: fp32 OIHW [Cout,Cin,3,3] -> (hi,lo) f16x2 [Cout, Kpad],
// K = ci*9 + dy*3 + dx.
// ---------------------------------------------------------------------------
__global__ void k_prep_w(const float* __restrict__ w, h2* __restrict__ q,
                         int Cout, int Cin, int Kpad) {
  int idx = blockIdx.x * TPB + threadIdx.x;
  int total = Cout * Kpad;
  if (idx >= total) return;
  int co = idx / Kpad;
  int k  = idx - co * Kpad;
  float v = 0.f;
  if (k < Cin * 9) v = w[(size_t)co * Cin * 9 + k];   // (co*Cin+ci)*9+tap
  f16 hi = (f16)v;
  h2 p; p.x = hi; p.y = (f16)(v - (float)hi);
  q[idx] = p;
}

// ---------------------------------------------------------------------------
// Activation prep: fp32 [BC,Hi,Wi] --(::sub subsample)--> zero-padded (hi,lo)
// f16x2 [BC, Hp, Wp] with 1-element zero border (Hp = Hi/sub + 2).
// ---------------------------------------------------------------------------
__global__ void k_prep_act(const float* __restrict__ x, h2* __restrict__ q,
                           int BC, int Hi, int Wi, int sub, int Hp, int Wp) {
  int idx = blockIdx.x * TPB + threadIdx.x;
  int total = BC * Hp * Wp;
  if (idx >= total) return;
  int bc = idx / (Hp * Wp);
  int r  = idx - bc * (Hp * Wp);
  int yy = r / Wp;
  int xx = r - yy * Wp;
  float v = 0.f;
  if (yy >= 1 && yy <= Hp - 2 && xx >= 1 && xx <= Wp - 2)
    v = x[(size_t)bc * Hi * Wi + (size_t)((yy - 1) * sub) * Wi + (size_t)((xx - 1) * sub)];
  f16 hi = (f16)v;
  h2 p; p.x = hi; p.y = (f16)(v - (float)hi);
  q[idx] = p;
}

// ---------------------------------------------------------------------------
// Implicit-GEMM 3x3 conv (pad 1, stride 1) with WMMA f16 hi/lo split:
// out = Ah*Bh + Ah*Bl + Al*Bh  (f32 accumulate) ~ fp32-class accuracy.
// One wave computes a 16(cout) x 64(pixel) tile = 4 WMMA N-tiles sharing the
// A (weight) fragments and the im2col address walk; K = Cin*9 in 32-chunks.
// mode 0: write fp32 NCHW (pre-instance-norm, ReLU applied).
// mode 1: write next layer's zero-padded (hi,lo) activation buffer (ReLU).
// ---------------------------------------------------------------------------
__global__ void __launch_bounds__(TPB)
k_conv_wmma(const h2* __restrict__ xq, const h2* __restrict__ wq,
            const float* __restrict__ bias,
            float* __restrict__ outF, h2* __restrict__ outQ,
            int B_, int Cin, int Cout, int H, int W,
            int Kreal, int Kpad, int mode)
{
  const int Hp = H + 2, Wp = W + 2;
  const int ntiles = (H * W) >> 6;            // 64-pixel groups per image
  const int per_img = (Cout >> 4) * ntiles;
  const int total = B_ * per_img;

  const int wave = threadIdx.x >> 5;
  int t = blockIdx.x * (TPB / 32) + wave;
  if (t >= total) return;                     // wave-uniform exit (EXEC full)

  const int lane = threadIdx.x & 31;
  const int n    = lane & 15;                 // B column / A row / D column
  const int half = lane >> 4;                 // K-half selector

  int b = t / per_img;          int r = t - b * per_img;
  int m = r / ntiles;           const int nt = r - m * ntiles;

  // 4 pixel groups of 16; group g covers linear pixels p0+16g .. p0+16g+15
  const int p0 = nt << 6;
  int yg[4], xg[4], dg[4];
#pragma unroll
  for (int g = 0; g < 4; ++g) {
    int pg = p0 + 16 * g;
    yg[g] = pg / W;
    xg[g] = pg - yg[g] * W;
  }
#pragma unroll
  for (int g = 0; g < 4; ++g)
    dg[g] = (yg[g] - yg[0]) * Wp + (xg[g] - xg[0]);   // lane-uniform offsets

  const h2* __restrict__ wr = wq + (size_t)(m * 16 + n) * Kpad;
  const h2* __restrict__ xb = xq + (size_t)b * Cin * Hp * Wp;
  const int ciStep = Hp * Wp - 3 * Wp;        // extra jump on dy-wrap

  v8f acc0 = {}, acc1 = {}, acc2 = {}, acc3 = {};
  const int Kmain = Kreal & ~31;

  for (int kc = 0; kc < Kmain; kc += 32) {
    v16h ah, al;
    // ---- A fragment: lanes 0-15 -> K kc+0..15; lanes 16-31 -> +16
    const int ka = kc + half * 16;
#pragma unroll
    for (int e = 0; e < 8; ++e) {
      h2 p0a = wr[ka + e];
      h2 p1a = wr[ka + 8 + e];
      ah[e] = p0a.x;  al[e] = p0a.y;
      ah[e + 8] = p1a.x;  al[e + 8] = p1a.y;
    }
    // ---- B fragments (4 groups): lane = column n, K = kc + half*16 + e
    const int k0 = kc + half * 16;
    int ci = k0 / 9;
    int tp = k0 - ci * 9;
    int dy = tp / 3;
    int dx = tp - dy * 3;
    int off = (ci * Hp + (yg[0] + dy)) * Wp + (xg[0] + n + dx);
    v16h bh0, bl0, bh1, bl1, bh2, bl2, bh3, bl3;
#pragma unroll
    for (int e = 0; e < 16; ++e) {
      h2 v0 = xb[off];
      h2 v1 = xb[off + dg[1]];
      h2 v2 = xb[off + dg[2]];
      h2 v3 = xb[off + dg[3]];
      bh0[e] = v0.x;  bl0[e] = v0.y;
      bh1[e] = v1.x;  bl1[e] = v1.y;
      bh2[e] = v2.x;  bl2[e] = v2.y;
      bh3[e] = v3.x;  bl3[e] = v3.y;
      // advance k -> k+1: dx++, wrap to (dy+1), wrap to (ci+1)
      ++dx; ++off;
      if (dx == 3) {
        dx = 0; ++dy; off += (Wp - 3);
        if (dy == 3) { dy = 0; off += ciStep; }
      }
    }
    acc0 = __builtin_amdgcn_wmma_f32_16x16x32_f16(false, ah, false, bh0, (short)0, acc0, false, false);
    acc0 = __builtin_amdgcn_wmma_f32_16x16x32_f16(false, ah, false, bl0, (short)0, acc0, false, false);
    acc0 = __builtin_amdgcn_wmma_f32_16x16x32_f16(false, al, false, bh0, (short)0, acc0, false, false);
    acc1 = __builtin_amdgcn_wmma_f32_16x16x32_f16(false, ah, false, bh1, (short)0, acc1, false, false);
    acc1 = __builtin_amdgcn_wmma_f32_16x16x32_f16(false, ah, false, bl1, (short)0, acc1, false, false);
    acc1 = __builtin_amdgcn_wmma_f32_16x16x32_f16(false, al, false, bh1, (short)0, acc1, false, false);
    acc2 = __builtin_amdgcn_wmma_f32_16x16x32_f16(false, ah, false, bh2, (short)0, acc2, false, false);
    acc2 = __builtin_amdgcn_wmma_f32_16x16x32_f16(false, ah, false, bl2, (short)0, acc2, false, false);
    acc2 = __builtin_amdgcn_wmma_f32_16x16x32_f16(false, al, false, bh2, (short)0, acc2, false, false);
    acc3 = __builtin_amdgcn_wmma_f32_16x16x32_f16(false, ah, false, bh3, (short)0, acc3, false, false);
    acc3 = __builtin_amdgcn_wmma_f32_16x16x32_f16(false, ah, false, bl3, (short)0, acc3, false, false);
    acc3 = __builtin_amdgcn_wmma_f32_16x16x32_f16(false, al, false, bh3, (short)0, acc3, false, false);
  }

  // ---- guarded tail chunk (only layer 1a: Kreal=18 < Kpad=32) ----
  if (Kmain < Kpad) {
    v16h ah, al;
    const int ka = Kmain + half * 16;
#pragma unroll
    for (int e = 0; e < 8; ++e) {
      h2 p0a = wr[ka + e];
      h2 p1a = wr[ka + 8 + e];
      ah[e] = p0a.x;  al[e] = p0a.y;
      ah[e + 8] = p1a.x;  al[e + 8] = p1a.y;
    }
    const int kb = Kmain + half * 16;
    v16h bh0, bl0, bh1, bl1, bh2, bl2, bh3, bl3;
#pragma unroll
    for (int e = 0; e < 16; ++e) {
      const int k = kb + e;
      f16 z = (f16)0.f;
      h2 v0, v1, v2, v3;
      v0.x = z; v0.y = z; v1 = v0; v2 = v0; v3 = v0;
      if (k < Kreal) {
        const int ci  = k / 9;
        const int tap = k - ci * 9;
        const int dy  = tap / 3;
        const int dx  = tap - dy * 3;
        const int off = (ci * Hp + (yg[0] + dy)) * Wp + (xg[0] + n + dx);
        v0 = xb[off];
        v1 = xb[off + dg[1]];
        v2 = xb[off + dg[2]];
        v3 = xb[off + dg[3]];
      }
      bh0[e] = v0.x;  bl0[e] = v0.y;
      bh1[e] = v1.x;  bl1[e] = v1.y;
      bh2[e] = v2.x;  bl2[e] = v2.y;
      bh3[e] = v3.x;  bl3[e] = v3.y;
    }
    acc0 = __builtin_amdgcn_wmma_f32_16x16x32_f16(false, ah, false, bh0, (short)0, acc0, false, false);
    acc0 = __builtin_amdgcn_wmma_f32_16x16x32_f16(false, ah, false, bl0, (short)0, acc0, false, false);
    acc0 = __builtin_amdgcn_wmma_f32_16x16x32_f16(false, al, false, bh0, (short)0, acc0, false, false);
    acc1 = __builtin_amdgcn_wmma_f32_16x16x32_f16(false, ah, false, bh1, (short)0, acc1, false, false);
    acc1 = __builtin_amdgcn_wmma_f32_16x16x32_f16(false, ah, false, bl1, (short)0, acc1, false, false);
    acc1 = __builtin_amdgcn_wmma_f32_16x16x32_f16(false, al, false, bh1, (short)0, acc1, false, false);
    acc2 = __builtin_amdgcn_wmma_f32_16x16x32_f16(false, ah, false, bh2, (short)0, acc2, false, false);
    acc2 = __builtin_amdgcn_wmma_f32_16x16x32_f16(false, ah, false, bl2, (short)0, acc2, false, false);
    acc2 = __builtin_amdgcn_wmma_f32_16x16x32_f16(false, al, false, bh2, (short)0, acc2, false, false);
    acc3 = __builtin_amdgcn_wmma_f32_16x16x32_f16(false, ah, false, bh3, (short)0, acc3, false, false);
    acc3 = __builtin_amdgcn_wmma_f32_16x16x32_f16(false, ah, false, bl3, (short)0, acc3, false, false);
    acc3 = __builtin_amdgcn_wmma_f32_16x16x32_f16(false, al, false, bh3, (short)0, acc3, false, false);
  }

  // ---- epilogue: D layout m = vgpr + 8*(lane>=16), n = lane&15 ----
#pragma unroll
  for (int rr = 0; rr < 8; ++rr) {
    const int co = m * 16 + half * 8 + rr;
    const float bv = bias[co];
    float vg[4];
    vg[0] = acc0[rr]; vg[1] = acc1[rr]; vg[2] = acc2[rr]; vg[3] = acc3[rr];
#pragma unroll
    for (int g = 0; g < 4; ++g) {
      float v = vg[g] + bv;
      v = v > 0.f ? v : 0.f;
      if (mode == 0) {
        outF[(((size_t)b * Cout + co) * H + yg[g]) * W + (xg[g] + n)] = v;
      } else {
        const size_t a = ((size_t)b * Cout + co) * (size_t)(Hp * Wp)
                       + (size_t)(yg[g] + 1) * Wp + (size_t)(xg[g] + n + 1);
        f16 hi = (f16)v;
        h2 p; p.x = hi; p.y = (f16)(v - (float)hi);
        outQ[a] = p;
      }
    }
  }
}

// ---------------------------------------------------------------------------
// InstanceNorm2d (biased var, eps=1e-5, no affine). One block per (b,c).
// ---------------------------------------------------------------------------
__global__ void __launch_bounds__(TPB)
k_inorm(const float* __restrict__ src, float* __restrict__ dst, int HW) {
  const int bc = blockIdx.x;
  const float* p = src + (size_t)bc * HW;
  float s = 0.f, s2 = 0.f;
  for (int i = threadIdx.x; i < HW; i += TPB) {
    float v = p[i];
    s += v; s2 += v * v;
  }
  __shared__ float sh[TPB], sh2[TPB];
  sh[threadIdx.x] = s; sh2[threadIdx.x] = s2;
  __syncthreads();
  for (int o = TPB / 2; o > 0; o >>= 1) {
    if (threadIdx.x < o) {
      sh[threadIdx.x]  += sh[threadIdx.x + o];
      sh2[threadIdx.x] += sh2[threadIdx.x + o];
    }
    __syncthreads();
  }
  const float mean = sh[0] / (float)HW;
  float var = sh2[0] / (float)HW - mean * mean;
  var = var > 0.f ? var : 0.f;
  const float rinv = rsqrtf(var + 1e-5f);
  float* q = dst + (size_t)bc * HW;
  for (int i = threadIdx.x; i < HW; i += TPB)
    q[i] = (p[i] - mean) * rinv;
}

// ---------------------------------------------------------------------------
// Edge-replicated pad: [BC,128,128] -> [BC,140,140] (pad 6, mode='edge')
// ---------------------------------------------------------------------------
__global__ void k_pad_edge(const float* __restrict__ x, float* __restrict__ o, int BC) {
  int idx = blockIdx.x * TPB + threadIdx.x;
  int total = BC * 140 * 140;
  if (idx >= total) return;
  int bc = idx / (140 * 140);
  int r  = idx - bc * (140 * 140);
  int yy = r / 140, xx = r - (r / 140) * 140;
  int ys = yy - 6; ys = ys < 0 ? 0 : (ys > 127 ? 127 : ys);
  int xs = xx - 6; xs = xs < 0 ? 0 : (xs > 127 ? 127 : xs);
  o[idx] = x[((size_t)bc * 128 + ys) * 128 + xs];
}

// ---------------------------------------------------------------------------
// align_1[b,c,y,x] = sum_{i,j in 13x13} corr[b, i*13+j, y*128+x] * pad[b,c,y+i,x+j]
// ---------------------------------------------------------------------------
__global__ void k_align(const float* __restrict__ pad, const float* __restrict__ corr,
                        float* __restrict__ out) {
  int idx = blockIdx.x * TPB + threadIdx.x;
  if (idx >= (4 << 21)) return;             // 4*128*128*128
  const int b = idx >> 21;
  const int r = idx & ((1 << 21) - 1);
  const int c = r >> 14;
  const int p = r & 16383;
  const int y = p >> 7, x = p & 127;
  const float* cp = corr + (size_t)b * 169 * 16384 + p;
  const float* pp = pad + ((size_t)(b * 128 + c) * 140 + y) * 140 + x;
  float acc = 0.f;
  for (int i = 0; i < 13; ++i) {
#pragma unroll
    for (int j = 0; j < 13; ++j)
      acc += cp[(size_t)(i * 13 + j) * 16384] * pp[(size_t)i * 140 + j];
  }
  out[idx] = acc;
}

// ---------------------------------------------------------------------------
extern "C" void kernel_launch(void* const* d_in, const int* in_sizes, int n_in,
                              void* d_out, int out_size, void* d_ws, size_t ws_size,
                              hipStream_t stream) {
  (void)in_sizes; (void)n_in; (void)out_size; (void)ws_size;
  const float* color = (const float*)d_in[0];
  const float* corr  = (const float*)d_in[1];
  const float* W8[8]; const float* Bi[8];
  for (int i = 0; i < 8; ++i) { W8[i] = (const float*)d_in[2 + 2 * i]; Bi[i] = (const float*)d_in[3 + 2 * i]; }
  float* out = (float*)d_out;
  char*  ws  = (char*)d_ws;

  static const int LCo[8] = {64, 64, 128, 128, 256, 256, 512, 512};
  static const int LCi[8] = {2,  64, 64,  128, 128, 256, 256, 512};
  static const int LKp[8] = {32, 576, 576, 1152, 1152, 2304, 2304, 4608};

  // ---- workspace layout: weight arena + two ping-pong activation arenas ----
  size_t woff[8]; size_t o = 0;
  for (int i = 0; i < 8; ++i) { woff[i] = o; o += (size_t)LCo[i] * LKp[i] * sizeof(h2); }
  const size_t OFF_A1 = (o + 255) & ~(size_t)255;
  const size_t SZ_A1  = 67108864;                 // max tenant: y1 (4*64*256*256*4)
  const size_t OFF_A2 = OFF_A1 + SZ_A1;           // max tenant: xp1b (68,161,536 B)
  h2* wq[8]; for (int i = 0; i < 8; ++i) wq[i] = (h2*)(ws + woff[i]);
  char* A1 = ws + OFF_A1;
  char* A2 = ws + OFF_A2;

  // ---- weight prep (8 layers) ----
  for (int i = 0; i < 8; ++i) {
    int total = LCo[i] * LKp[i];
    k_prep_w<<<(total + TPB - 1) / TPB, TPB, 0, stream>>>(W8[i], wq[i], LCo[i], LCi[i], LKp[i]);
  }

  auto conv = [&](const h2* xq, int L, float* outF, h2* outQ, int H, int W, int mode) {
    int tiles = 4 * (LCo[L] / 16) * ((H * W) / 64);
    k_conv_wmma<<<(tiles + 7) / 8, TPB, 0, stream>>>(
        xq, wq[L], Bi[L], outF, outQ, 4, LCi[L], LCo[L], H, W, LCi[L] * 9, LKp[L], mode);
  };

  // ---- block 1: color[::2,::2] -> conv1a -> conv1b -> inorm ----
  h2* xp1a = (h2*)A1;
  { int total = 8 * 258 * 258;
    k_prep_act<<<(total + TPB - 1) / TPB, TPB, 0, stream>>>(color, xp1a, 8, 512, 512, 2, 258, 258); }
  h2* xp1b = (h2*)A2;
  hipMemsetAsync(xp1b, 0, (size_t)4 * 64 * 258 * 258 * sizeof(h2), stream);
  conv(xp1a, 0, nullptr, xp1b, 256, 256, 1);
  float* y1 = (float*)A1;
  conv(xp1b, 1, y1, nullptr, 256, 256, 0);
  k_inorm<<<4 * 64, TPB, 0, stream>>>(y1, y1, 256 * 256);

  // ---- block 2 ----
  h2* xp2a = (h2*)A2;
  { int total = 4 * 64 * 130 * 130;
    k_prep_act<<<(total + TPB - 1) / TPB, TPB, 0, stream>>>(y1, xp2a, 4 * 64, 256, 256, 2, 130, 130); }
  h2* xp2b = (h2*)A1;
  hipMemsetAsync(xp2b, 0, (size_t)4 * 128 * 130 * 130 * sizeof(h2), stream);
  conv(xp2a, 2, nullptr, xp2b, 128, 128, 1);
  float* y2 = (float*)A2;
  conv(xp2b, 3, y2, nullptr, 128, 128, 0);
  k_inorm<<<4 * 128, TPB, 0, stream>>>(y2, y2, 128 * 128);

  // ---- correlation aggregation -> align_1 (output 0) ----
  float* pad2 = (float*)A1;
  { int total = 4 * 128 * 140 * 140;
    k_pad_edge<<<(total + TPB - 1) / TPB, TPB, 0, stream>>>(y2, pad2, 4 * 128); }
  k_align<<<(8388608 + TPB - 1) / TPB, TPB, 0, stream>>>(pad2, corr, out);

  // ---- block 3: align_1[::2,::2] -> conv3a -> conv3b -> inorm -> align_2 ----
  h2* xp3a = (h2*)A2;
  { int total = 4 * 128 * 66 * 66;
    k_prep_act<<<(total + TPB - 1) / TPB, TPB, 0, stream>>>(out, xp3a, 4 * 128, 128, 128, 2, 66, 66); }
  h2* xp3b = (h2*)A1;
  hipMemsetAsync(xp3b, 0, (size_t)4 * 256 * 66 * 66 * sizeof(h2), stream);
  conv(xp3a, 4, nullptr, xp3b, 64, 64, 1);
  float* y3 = (float*)A2;
  conv(xp3b, 5, y3, nullptr, 64, 64, 0);
  float* out2 = out + 8388608;
  k_inorm<<<4 * 256, TPB, 0, stream>>>(y3, out2, 64 * 64);

  // ---- block 4: align_2[::2,::2] -> conv4a -> conv4b -> inorm -> align_3 ----
  h2* xp4a = (h2*)A1;
  { int total = 4 * 256 * 34 * 34;
    k_prep_act<<<(total + TPB - 1) / TPB, TPB, 0, stream>>>(out2, xp4a, 4 * 256, 64, 64, 2, 34, 34); }
  h2* xp4b = (h2*)A2;
  hipMemsetAsync(xp4b, 0, (size_t)4 * 512 * 34 * 34 * sizeof(h2), stream);
  conv(xp4a, 6, nullptr, xp4b, 32, 32, 1);
  float* y4 = (float*)A1;
  conv(xp4b, 7, y4, nullptr, 32, 32, 0);
  float* out3 = out2 + 4194304;
  k_inorm<<<4 * 512, TPB, 0, stream>>>(y4, out3, 32 * 32);
}
